// FusionHead_39522289058128
// MI455X (gfx1250) — matrix operand order repaired and checked by reading, not compiled, not run on hardware
//
#include <hip/hip_runtime.h>
#include <math.h>

typedef _Float16 v16h __attribute__((ext_vector_type(16)));
typedef float    v8f  __attribute__((ext_vector_type(8)));

#define DDIM   771
#define EDIM   768
#define HDIM   96
#define CDIM   3
#define KPAD   800          // 25 K-tiles of 32
#define ASTRIDE 808         // f16 per row (404 DWORDs, conflict-free vs 64 banks)
#define HSTRIDE 100
#define ROWS_PER_BLOCK 64

// ---------------------------------------------------------------------------
// Prelude: pack w1 [771][96] f32 (row-major, x@w1) into transposed padded f16
// w1T[96][808]: w1T[h*808 + k] = (k < 771) ? w1[k*96 + h] : 0
// ---------------------------------------------------------------------------
__global__ void pack_w1T(const float* __restrict__ w1, _Float16* __restrict__ w1T) {
  int idx = blockIdx.x * 256 + threadIdx.x;
  if (idx >= HDIM * ASTRIDE) return;
  int h = idx / ASTRIDE;
  int k = idx - h * ASTRIDE;
  float v = (k < DDIM) ? w1[k * HDIM + h] : 0.0f;
  w1T[h * ASTRIDE + k] = (_Float16)v;
}

// ---------------------------------------------------------------------------
// Main fused kernel: 256 threads = 8 wave32, 64 rows per block.
// ---------------------------------------------------------------------------
__launch_bounds__(256)
__global__ void fusion_head_kernel(const float* __restrict__ emb,
                                   const float* __restrict__ tp,
                                   const float* __restrict__ gamma,
                                   const float* __restrict__ beta,
                                   const _Float16* __restrict__ w1T,
                                   const float* __restrict__ b1,
                                   const float* __restrict__ w2,
                                   const float* __restrict__ b2,
                                   float* __restrict__ out) {
  __shared__ __align__(16) _Float16 Alds[ROWS_PER_BLOCK * ASTRIDE]; // 103,424 B
  __shared__ float Hlds[ROWS_PER_BLOCK * HSTRIDE];                  //  25,600 B

  const int tid  = threadIdx.x;
  const int lane = tid & 31;
  const int wave = tid >> 5;
  const long long rowbase = (long long)blockIdx.x * ROWS_PER_BLOCK;

  // ---- preload gamma/beta slices for this lane (L2 resident, once per block)
  float4 g4[6], bb4[6];
#pragma unroll
  for (int j = 0; j < 6; ++j) {
    int e = (lane + 32 * j) * 4;
    g4[j]  = *(const float4*)(gamma + e);
    bb4[j] = *(const float4*)(beta  + e);
  }
  float gt = 0.0f, bt = 0.0f;
  if (lane < CDIM) { gt = gamma[EDIM + lane]; bt = beta[EDIM + lane]; }

  // ---- Phase 1: stream rows, layernorm in registers, store f16 to LDS -----
  for (int i = 0; i < 8; ++i) {
    const int r = wave * 8 + i;                 // local row 0..63
    const long long grow = rowbase + r;
    const float* ep = emb + grow * EDIM;

    float4 x[6];
    float s = 0.0f, s2 = 0.0f;
#pragma unroll
    for (int j = 0; j < 6; ++j) {
      x[j] = *(const float4*)(ep + (lane + 32 * j) * 4);
      s  += x[j].x + x[j].y + x[j].z + x[j].w;
      s2 += x[j].x * x[j].x + x[j].y * x[j].y + x[j].z * x[j].z + x[j].w * x[j].w;
    }
    float tv = (lane < CDIM) ? tp[grow * CDIM + lane] : 0.0f;
    s += tv; s2 += tv * tv;

#pragma unroll
    for (int m = 16; m >= 1; m >>= 1) {
      s  += __shfl_xor(s,  m, 32);
      s2 += __shfl_xor(s2, m, 32);
    }
    const float mu  = s * (1.0f / (float)DDIM);
    const float var = s2 * (1.0f / (float)DDIM) - mu * mu;
    const float rs  = rsqrtf(var + 1e-5f);

    _Float16* arow = &Alds[r * ASTRIDE];
#pragma unroll
    for (int j = 0; j < 6; ++j) {
      union { _Float16 h[4]; uint2 u; } p;
      p.h[0] = (_Float16)((x[j].x - mu) * rs * g4[j].x + bb4[j].x);
      p.h[1] = (_Float16)((x[j].y - mu) * rs * g4[j].y + bb4[j].y);
      p.h[2] = (_Float16)((x[j].z - mu) * rs * g4[j].z + bb4[j].z);
      p.h[3] = (_Float16)((x[j].w - mu) * rs * g4[j].w + bb4[j].w);
      *(uint2*)(arow + (lane + 32 * j) * 4) = p.u;
    }
    // tree probs (k=768..770) + zero pad up to 800
    float tn = (lane < CDIM) ? ((tv - mu) * rs * gt + bt) : 0.0f;
    float a0 = __shfl(tn, 0, 32);
    float a1 = __shfl(tn, 1, 32);
    float a2 = __shfl(tn, 2, 32);
    if (lane == 0) {
      union { _Float16 h[2]; unsigned u; } q;
      q.h[0] = (_Float16)a0; q.h[1] = (_Float16)a1;
      *(unsigned*)(arow + 768) = q.u;
    } else if (lane == 1) {
      union { _Float16 h[2]; unsigned u; } q;
      q.h[0] = (_Float16)a2; q.h[1] = (_Float16)0.0f;
      *(unsigned*)(arow + 770) = q.u;
    } else if (lane < 20) {
      *(unsigned*)(arow + 768 + 2 * lane) = 0u;   // k = 772..807 zeroed
    }
  }
  __syncthreads();

  // ---- Phase 2: WMMA GEMM1  (x[64x800] @ w1T^T[800x96]) -------------------
  // waves 0-3: M-tiles 0-3, N-tiles 0-2 ; waves 4-7: M-tiles 0-3, N-tiles 3-5
  const int mt     = wave & 3;
  const int nh     = wave >> 2;
  const int nrow   = lane & 15;
  const int khalf  = lane >> 4;
  const int mlocal = mt * 16 + nrow;

  v8f acc[3];
#pragma unroll
  for (int n = 0; n < 3; ++n) acc[n] = (v8f)0.0f;

  const _Float16* Arow = &Alds[mlocal * ASTRIDE];
  for (int kt = 0; kt < 25; ++kt) {
    // A fragment: 16-bit A 16x32 layout -> per lane K {kb..kb+7} and {kb+16..kb+23}
    const int ka = kt * 32 + khalf * 8;
    union { v16h v; uint4 u[2]; } af;
    af.u[0] = *(const uint4*)(Arow + ka);
    af.u[1] = *(const uint4*)(Arow + ka + 16);

    const int kb = kt * 32 + khalf * 8;         // B: per lane K {kb..kb+7} and {kb+16..kb+23}
#pragma unroll
    for (int nn = 0; nn < 3; ++nn) {
      const int ng = (nh * 3 + nn) * 16 + nrow;
      const _Float16* bp = w1T + ng * ASTRIDE + kb;
      union { v16h v; uint4 u[2]; } bf;
      bf.u[0] = *(const uint4*)(bp);
      bf.u[1] = *(const uint4*)(bp + 16);
      acc[nn] = __builtin_amdgcn_wmma_f32_16x16x32_f16(
          false, af.v, false, bf.v, (short)0, acc[nn], false, false);
    }
  }

  // ---- Epilogue: +b1, exact GELU, stage to LDS ----------------------------
#pragma unroll
  for (int nn = 0; nn < 3; ++nn) {
    const int ng = (nh * 3 + nn) * 16 + nrow;
    const float bias = b1[ng];
#pragma unroll
    for (int r = 0; r < 8; ++r) {
      const int m = mt * 16 + r + 8 * khalf;    // C/D layout: V r -> M=r / r+8
      float h = acc[nn][r] + bias;
      float g = 0.5f * h * (1.0f + erff(h * 0.70710678118654752f));
      Hlds[m * HSTRIDE + ng] = g;
    }
  }
  __syncthreads();

  // ---- GEMM2: [64x96] @ [96x3] + b2 ---------------------------------------
  if (tid < ROWS_PER_BLOCK * CDIM) {
    const int rr = tid / CDIM;
    const int c  = tid - rr * CDIM;
    float sum = b2[c];
    const float* hr = &Hlds[rr * HSTRIDE];
#pragma unroll 8
    for (int h = 0; h < HDIM; ++h) sum += hr[h] * w2[h * CDIM + c];
    out[(rowbase + rr) * CDIM + c] = sum;
  }
}

// ---------------------------------------------------------------------------
extern "C" void kernel_launch(void* const* d_in, const int* in_sizes, int n_in,
                              void* d_out, int out_size, void* d_ws, size_t ws_size,
                              hipStream_t stream) {
  const float* emb   = (const float*)d_in[0];
  const float* tp    = (const float*)d_in[1];
  const float* gamma = (const float*)d_in[2];
  const float* beta  = (const float*)d_in[3];
  const float* w1    = (const float*)d_in[4];
  const float* b1    = (const float*)d_in[5];
  const float* w2    = (const float*)d_in[6];
  const float* b2    = (const float*)d_in[7];
  float* out = (float*)d_out;
  _Float16* w1T = (_Float16*)d_ws;              // 96*808*2 = 155,136 B

  const int B = in_sizes[0] / EDIM;             // 131072
  const int packN = HDIM * ASTRIDE;
  pack_w1T<<<(packN + 255) / 256, 256, 0, stream>>>(w1, w1T);
  fusion_head_kernel<<<B / ROWS_PER_BLOCK, 256, 0, stream>>>(
      emb, tp, gamma, beta, w1T, b1, w2, b2, out);
}
